// Baseline_mb_24189255811183
// MI455X (gfx1250) — matrix operand order repaired and checked by reading, not compiled, not run on hardware
//
#include <hip/hip_runtime.h>
#include <cmath>

typedef __attribute__((ext_vector_type(16))) _Float16 v16h;
typedef __attribute__((ext_vector_type(8)))  float    v8f;

constexpr int cP = 16384;  // paths
constexpr int cT = 8;      // path length
constexpr int cL = 4096;   // links
constexpr int cK = 16;     // paths per link
constexpr int cN = 2048;   // nodes
constexpr int cK2 = 32;    // paths per node
constexpr int cM = 8;      // links per node
constexpr int cD = 64;     // hidden dim
constexpr int cITER = 8;

#define WMMA_F16(A,B,C) __builtin_amdgcn_wmma_f32_16x16x32_f16(false,(A),false,(B),(short)0,(C),false,false)

__device__ __forceinline__ float frcp(float x){ return __builtin_amdgcn_rcpf(x); }
__device__ __forceinline__ float sigmf(float x){ return frcp(1.0f + __expf(-x)); }
__device__ __forceinline__ float tanh_fast(float x){ return 1.0f - 2.0f*frcp(1.0f + __expf(2.0f*x)); }

// ---- A fragment (16x32 f16) from f32 row-major via float4 loads ----
// lanes 0-15 = rows, K={0..7,16..23}; lanes 16-31 same rows, K={8..15,24..31}
__device__ __forceinline__ v16h load_a_f32(const float* __restrict__ base, int ld, int lane){
  int m = lane & 15, half = lane >> 4;
  const float4* rp = (const float4*)(base + (size_t)m * ld);
  float4 x0 = rp[half*2], x1 = rp[half*2+1];
  float4 y0 = rp[4+half*2], y1 = rp[5+half*2];
  v16h a;
  a[0]=(_Float16)x0.x; a[1]=(_Float16)x0.y; a[2]=(_Float16)x0.z; a[3]=(_Float16)x0.w;
  a[4]=(_Float16)x1.x; a[5]=(_Float16)x1.y; a[6]=(_Float16)x1.z; a[7]=(_Float16)x1.w;
  a[8]=(_Float16)y0.x; a[9]=(_Float16)y0.y; a[10]=(_Float16)y0.z; a[11]=(_Float16)y0.w;
  a[12]=(_Float16)y1.x; a[13]=(_Float16)y1.y; a[14]=(_Float16)y1.z; a[15]=(_Float16)y1.w;
  return a;
}

// A fragment = sum of two gathered rows (per-lane row pointers, k-offset included)
__device__ __forceinline__ v16h load_a_sum(const float* __restrict__ pa,
                                           const float* __restrict__ pb, int lane){
  int half = lane >> 4;
  const float4* ra = (const float4*)pa;
  const float4* rb = (const float4*)pb;
  float4 xa0 = ra[half*2], xa1 = ra[half*2+1], ya0 = ra[4+half*2], ya1 = ra[5+half*2];
  float4 xb0 = rb[half*2], xb1 = rb[half*2+1], yb0 = rb[4+half*2], yb1 = rb[5+half*2];
  v16h a;
  a[0]=(_Float16)(xa0.x+xb0.x); a[1]=(_Float16)(xa0.y+xb0.y);
  a[2]=(_Float16)(xa0.z+xb0.z); a[3]=(_Float16)(xa0.w+xb0.w);
  a[4]=(_Float16)(xa1.x+xb1.x); a[5]=(_Float16)(xa1.y+xb1.y);
  a[6]=(_Float16)(xa1.z+xb1.z); a[7]=(_Float16)(xa1.w+xb1.w);
  a[8]=(_Float16)(ya0.x+yb0.x); a[9]=(_Float16)(ya0.y+yb0.y);
  a[10]=(_Float16)(ya0.z+yb0.z); a[11]=(_Float16)(ya0.w+yb0.w);
  a[12]=(_Float16)(ya1.x+yb1.x); a[13]=(_Float16)(ya1.y+yb1.y);
  a[14]=(_Float16)(ya1.z+yb1.z); a[15]=(_Float16)(ya1.w+yb1.w);
  return a;
}

// ---- Weight prepack: f32 (K x Ncols) row-major -> f16 B-fragment blocks ----
// fragment index = kt*(Ncols/16)+ct; lane's 16 halves stored contiguously -> b128 load
__global__ void k_prepack(const float* __restrict__ W, int Ncols, int total,
                          _Float16* __restrict__ out){
  int i = blockIdx.x*blockDim.x + threadIdx.x;
  if (i >= total) return;                   // total = (K/32)*(Ncols/16)*32
  int lane = i & 31;
  int tile = i >> 5;
  int NT = Ncols >> 4;
  int ct = tile % NT, kt = tile / NT;
  int n = lane & 15, half = lane >> 4;
  v16h b;
#pragma unroll
  for (int v = 0; v < 8; ++v){
    int k = kt*32 + (v>>2)*16 + half*8 + 2*(v&3);
    b[2*v]   = (_Float16)W[(size_t)k * Ncols + ct*16 + n];
    b[2*v+1] = (_Float16)W[(size_t)(k+1) * Ncols + ct*16 + n];
  }
  ((v16h*)out)[i] = b;
}

__device__ __forceinline__ v16h load_b_pk(const _Float16* __restrict__ Wp, int tile, int lane){
  return ((const v16h*)Wp)[(size_t)tile*32 + lane];
}

// ---- Generic GEMM: C = act(A @ W + bias); one wave per 16 rows ----
template<int NT, int KT, int ACT>
__global__ void __launch_bounds__(32, 1)
gemm_bias_act(const float* __restrict__ A, int ldA,
              const _Float16* __restrict__ Wp,
              const float* __restrict__ bias,
              float* __restrict__ C, int ldC, int rows){
  int r0 = blockIdx.x * 16;
  if (r0 >= rows) return;
  int lane = threadIdx.x;
  int n = lane & 15, half = lane >> 4;
  __builtin_prefetch(Wp, 0, 3);
  v8f acc[NT];
#pragma unroll
  for (int ct = 0; ct < NT; ++ct){ v8f z = {}; acc[ct] = z; }
#pragma unroll
  for (int kt = 0; kt < KT; ++kt){
    v16h a = load_a_f32(A + (size_t)r0 * ldA + kt*32, ldA, lane);
    v16h bf[NT];
#pragma unroll
    for (int ct = 0; ct < NT; ++ct) bf[ct] = load_b_pk(Wp, kt*NT + ct, lane);
#pragma unroll
    for (int ct = 0; ct < NT; ++ct) acc[ct] = WMMA_F16(a, bf[ct], acc[ct]);
  }
#pragma unroll
  for (int ct = 0; ct < NT; ++ct){
    float bv = bias[ct*16 + n];
#pragma unroll
    for (int r = 0; r < 8; ++r){
      float v = acc[ct][r] + bv;
      if (ACT) v = fmaxf(v, 0.0f);
      C[(size_t)(r0 + half*8 + r) * ldC + ct*16 + n] = v;
    }
  }
}

// ---- Fused GRU step: h' = z*h + (1-z)*tanh(xc + r*hc); one wave per 16 rows ----
// Per 16-col group: 4 live accumulators; B fragments preloaded in two kt-halves
// (6 fragments each) to bound register pressure; 4 interleaved WMMA chains; group
// retired with its GRU epilogue before the next group.
// mode 0 (paths): x row = link_state[ltp[p,t]] + device_state[ntp[p,t]], also writes pss[:,t+1]
// mode 1 (links/devices): x row = xsrc[row]
__global__ void __launch_bounds__(32, 1)
gru_fused(const float* __restrict__ xsrc,
          const int* __restrict__ ltp, const int* __restrict__ ntp,
          const float* __restrict__ ls, const float* __restrict__ ds,
          int t,
          const _Float16* __restrict__ pWx, const _Float16* __restrict__ pWh,
          const float* __restrict__ bx, const float* __restrict__ bh,
          float* __restrict__ h, float* __restrict__ pss,
          int rows, int mode){
  int r0 = blockIdx.x * 16;
  if (r0 >= rows) return;
  int lane = threadIdx.x;
  int n = lane & 15, half = lane >> 4;
  __builtin_prefetch(pWx, 0, 3);
  __builtin_prefetch(pWh, 0, 3);

  // A fragments, loaded once (old h read completes before any h write below)
  v16h ax0, ax1, ah0, ah1;
  if (mode == 0){
    int p  = r0 + (lane & 15);
    int li = ltp[p*cT + t];
    int ni = ntp[p*cT + t];
    const float* pl = ls + (size_t)li * cD;
    const float* pd = ds + (size_t)ni * cD;
    ax0 = load_a_sum(pl,      pd,      lane);
    ax1 = load_a_sum(pl + 32, pd + 32, lane);
  } else {
    ax0 = load_a_f32(xsrc + (size_t)r0 * cD,      cD, lane);
    ax1 = load_a_f32(xsrc + (size_t)r0 * cD + 32, cD, lane);
  }
  ah0 = load_a_f32(h + (size_t)r0 * cD,      cD, lane);
  ah1 = load_a_f32(h + (size_t)r0 * cD + 32, cD, lane);

#pragma unroll
  for (int ct = 0; ct < 4; ++ct){
    v8f accZ = {}, accR = {}, accXc = {}, accHc = {};
    // kt = 0 half: 6 B fragments, 6 WMMAs (4 independent chains interleaved)
    {
      v16h bxz = load_b_pk(pWx,      ct, lane);
      v16h bhz = load_b_pk(pWh,      ct, lane);
      v16h bxr = load_b_pk(pWx,  4 + ct, lane);
      v16h bhr = load_b_pk(pWh,  4 + ct, lane);
      v16h bxc = load_b_pk(pWx,  8 + ct, lane);
      v16h bhc = load_b_pk(pWh,  8 + ct, lane);
      accZ  = WMMA_F16(ax0, bxz, accZ);
      accR  = WMMA_F16(ax0, bxr, accR);
      accXc = WMMA_F16(ax0, bxc, accXc);
      accHc = WMMA_F16(ah0, bhc, accHc);
      accZ  = WMMA_F16(ah0, bhz, accZ);
      accR  = WMMA_F16(ah0, bhr, accR);
    }
    // kt = 1 half
    {
      v16h bxz = load_b_pk(pWx, 12 + ct, lane);
      v16h bhz = load_b_pk(pWh, 12 + ct, lane);
      v16h bxr = load_b_pk(pWx, 16 + ct, lane);
      v16h bhr = load_b_pk(pWh, 16 + ct, lane);
      v16h bxc = load_b_pk(pWx, 20 + ct, lane);
      v16h bhc = load_b_pk(pWh, 20 + ct, lane);
      accXc = WMMA_F16(ax1, bxc, accXc);
      accHc = WMMA_F16(ah1, bhc, accHc);
      accZ  = WMMA_F16(ax1, bxz, accZ);
      accR  = WMMA_F16(ax1, bxr, accR);
      accZ  = WMMA_F16(ah1, bhz, accZ);
      accR  = WMMA_F16(ah1, bhr, accR);
    }

    // GRU epilogue for cols [ct*16, ct*16+16)
    int j = ct*16 + n;
    float bz  = bx[j]       + bh[j];
    float br  = bx[64 + j]  + bh[64 + j];
    float bxcv = bx[128 + j];
    float bhcv = bh[128 + j];
#pragma unroll
    for (int r = 0; r < 8; ++r){
      int row = r0 + half*8 + r;
      float hv = h[(size_t)row * cD + j];
      float z  = sigmf(accZ[r] + bz);
      float rg = sigmf(accR[r] + br);
      float c  = tanh_fast(accXc[r] + bxcv + rg * (accHc[r] + bhcv));
      float hn = z*hv + (1.0f - z)*c;
      h[(size_t)row * cD + j] = hn;
      if (pss) pss[((size_t)row * 9 + t + 1) * cD + j] = hn;
    }
  }
}

// ---- elementwise / gather kernels ----
__global__ void k_load(const float* __restrict__ ft, const float* __restrict__ cap,
                       const int* __restrict__ ptl, float* __restrict__ load_b){
  int l = blockIdx.x*blockDim.x + threadIdx.x;
  if (l >= cL) return;
  float s = 0.f;
  for (int k = 0; k < cK; ++k) s += ft[ptl[(l*cK + k)*2]];
  load_b[l] = s / (cap[l] * 1.0e9f);
}

__global__ void k_path_l1(const float* ft, const float* fp, const float* fs,
                          const float* w1, const float* b1, float* out){
  int i = blockIdx.x*blockDim.x + threadIdx.x;
  if (i >= cP*cD) return;
  int p = i / cD, d = i % cD;
  float f0 = ft[p]*1e-4f, f1 = fp[p]*1e-3f, f2 = fs[p]*1e-3f;
  out[i] = fmaxf(f0*w1[d] + f1*w1[cD+d] + f2*w1[2*cD+d] + b1[d], 0.f);
}

__global__ void k_link_l1(const float* cap, const float* load_b, const int* ldt,
                          const float* w1, const float* b1, float* out){
  int i = blockIdx.x*blockDim.x + threadIdx.x;
  if (i >= cL*cD) return;
  int l = i / cD, d = i % cD;
  float f0 = cap[l]*1e-2f, f1 = load_b[l], f2 = (ldt[l]==0) ? 1.f : 0.f;
  out[i] = fmaxf(f0*w1[d] + f1*w1[cD+d] + f2*w1[2*cD+d] + b1[d], 0.f);
}

__global__ void k_dlm(const float* lst, const int* ltn, float* dlm){
  int nd = blockIdx.x*blockDim.x + threadIdx.x;
  if (nd >= cN) return;
  float s = 0.f;
  for (int m = 0; m < cM; ++m){
    const float4* row = (const float4*)(lst + (size_t)ltn[nd*cM + m] * cD);
#pragma unroll
    for (int d = 0; d < cD/4; ++d){
      float4 v = row[d];
      s += v.x + v.y + v.z + v.w;
    }
  }
  dlm[nd] = s / (float)cD;
}

__global__ void k_dev_l1(const int* nodes, const float* dlm, const float* w1,
                         const float* b1, float* out){
  int i = blockIdx.x*blockDim.x + threadIdx.x;
  if (i >= cN*cD) return;
  int nd = i / cD, d = i % cD;
  float f0 = (nodes[nd]==0) ? 1.f : 0.f;
  out[i] = fmaxf(f0*w1[d] + dlm[nd]*w1[cD+d] + b1[d], 0.f);
}

__global__ void k_copy_prev(const float* h, float* pss){
  int i = blockIdx.x*blockDim.x + threadIdx.x;
  if (i >= cP*cD) return;
  int p = i / cD, d = i % cD;
  pss[((size_t)p*9)*cD + d] = h[i];
}

__global__ void k_agg(const float* pss, const int* ptx, int Kc, float* agg, int rows){
  int i = blockIdx.x*blockDim.x + threadIdx.x;
  if (i >= rows*cD) return;
  int rr = i / cD, d = i % cD;
  float s = 0.f;
  for (int k = 0; k < Kc; ++k){
    int p    = ptx[(rr*Kc + k)*2];
    int sidx = ptx[(rr*Kc + k)*2 + 1];
    s += pss[((size_t)p*9 + sidx)*cD + d];
  }
  agg[i] = s;
}

__global__ void k_pssc(const float* pss, float* out){
  int i = blockIdx.x*blockDim.x + threadIdx.x;
  if (i >= cP*cT*cD) return;
  int p = i / (cT*cD); int rem = i % (cT*cD); int t = rem / cD; int d = rem % cD;
  out[i] = pss[((size_t)p*9 + 1 + t)*cD + d];
}

__global__ void k_final(const float* h2, const float* w3, const float* b3,
                        const float* cap, const int* ltp, float* out){
  int p = blockIdx.x*blockDim.x + threadIdx.x;
  if (p >= cP) return;
  float s = 0.f;
  for (int t = 0; t < cT; ++t){
    const float* row = h2 + (size_t)(p*cT + t) * 16;
    float v = b3[0];
#pragma unroll
    for (int i = 0; i < 16; ++i) v += row[i]*w3[i];
    float sp = (v > 20.f) ? v : log1pf(__expf(v));
    s += sp / cap[ltp[p*cT + t]];
  }
  out[p] = s;
}

extern "C" void kernel_launch(void* const* d_in, const int* in_sizes, int n_in,
                              void* d_out, int out_size, void* d_ws, size_t ws_size,
                              hipStream_t stream){
  const float* ft   = (const float*)d_in[0];
  const float* fpk  = (const float*)d_in[1];
  const float* fsz  = (const float*)d_in[2];
  const float* cap  = (const float*)d_in[3];
  const float* pe_w1=(const float*)d_in[4];  const float* pe_b1=(const float*)d_in[5];
  const float* pe_w2=(const float*)d_in[6];  const float* pe_b2=(const float*)d_in[7];
  const float* le_w1=(const float*)d_in[8];  const float* le_b1=(const float*)d_in[9];
  const float* le_w2=(const float*)d_in[10]; const float* le_b2=(const float*)d_in[11];
  const float* de_w1=(const float*)d_in[12]; const float* de_b1=(const float*)d_in[13];
  const float* de_w2=(const float*)d_in[14]; const float* de_b2=(const float*)d_in[15];
  const float* pgru_wx=(const float*)d_in[16]; const float* pgru_wh=(const float*)d_in[17];
  const float* pgru_bx=(const float*)d_in[18]; const float* pgru_bh=(const float*)d_in[19];
  const float* lgru_wx=(const float*)d_in[20]; const float* lgru_wh=(const float*)d_in[21];
  const float* lgru_bx=(const float*)d_in[22]; const float* lgru_bh=(const float*)d_in[23];
  const float* dgru_wx=(const float*)d_in[24]; const float* dgru_wh=(const float*)d_in[25];
  const float* dgru_bx=(const float*)d_in[26]; const float* dgru_bh=(const float*)d_in[27];
  const float* ro_w1=(const float*)d_in[28]; const float* ro_b1=(const float*)d_in[29];
  const float* ro_w2=(const float*)d_in[30]; const float* ro_b2=(const float*)d_in[31];
  const float* ro_w3=(const float*)d_in[32]; const float* ro_b3=(const float*)d_in[33];
  const int* ltp   = (const int*)d_in[34];
  const int* ntp   = (const int*)d_in[35];
  const int* ptl   = (const int*)d_in[36];
  const int* ptn   = (const int*)d_in[37];
  const int* ltn   = (const int*)d_in[38];
  const int* nodes = (const int*)d_in[39];
  const int* ldt   = (const int*)d_in[40];
  float* out = (float*)d_out;

  float* w = (float*)d_ws;
  size_t o = 0;
  float* load_b = w + o; o += cL;
  float* dlm    = w + o; o += cN;
  float* hP     = w + o; o += (size_t)cP*cD;        // path_state
  float* hL     = w + o; o += (size_t)cL*cD;        // link_state
  float* hD     = w + o; o += (size_t)cN*cD;        // device_state
  float* pss    = w + o; o += (size_t)cP*9*cD;      // [P,9,D]
  float* agg    = w + o; o += (size_t)cL*cD;        // link/device aggregation
  float* s1     = w + o; o += (size_t)cP*cT*cD;     // scratch: enc L1 / pss contiguous
  float* s2     = w + o; o += (size_t)cP*cT*32;     // readout h1
  float* s3     = w + o; o += (size_t)cP*cT*16;     // readout h2

  // prepacked f16 weights (element count == f32 element count)
  _Float16* h16 = (_Float16*)(w + o);
  size_t q = 0;
  _Float16* ppe_w2 = h16 + q; q += 64*64;
  _Float16* ple_w2 = h16 + q; q += 64*64;
  _Float16* pde_w2 = h16 + q; q += 64*64;
  _Float16* ppg_wx = h16 + q; q += 64*192;
  _Float16* ppg_wh = h16 + q; q += 64*192;
  _Float16* plg_wx = h16 + q; q += 64*192;
  _Float16* plg_wh = h16 + q; q += 64*192;
  _Float16* pdg_wx = h16 + q; q += 64*192;
  _Float16* pdg_wh = h16 + q; q += 64*192;
  _Float16* pro_w1 = h16 + q; q += 64*32;
  _Float16* pro_w2 = h16 + q; q += 32*16;
  (void)ws_size; (void)n_in; (void)in_sizes; (void)out_size;

  // ---- prepack weights into WMMA B-fragment order (f16) ----
  k_prepack<<<1, 256, 0, stream>>>(pe_w2, 64, 2*4*32, ppe_w2);
  k_prepack<<<1, 256, 0, stream>>>(le_w2, 64, 2*4*32, ple_w2);
  k_prepack<<<1, 256, 0, stream>>>(de_w2, 64, 2*4*32, pde_w2);
  k_prepack<<<3, 256, 0, stream>>>(pgru_wx, 192, 2*12*32, ppg_wx);
  k_prepack<<<3, 256, 0, stream>>>(pgru_wh, 192, 2*12*32, ppg_wh);
  k_prepack<<<3, 256, 0, stream>>>(lgru_wx, 192, 2*12*32, plg_wx);
  k_prepack<<<3, 256, 0, stream>>>(lgru_wh, 192, 2*12*32, plg_wh);
  k_prepack<<<3, 256, 0, stream>>>(dgru_wx, 192, 2*12*32, pdg_wx);
  k_prepack<<<3, 256, 0, stream>>>(dgru_wh, 192, 2*12*32, pdg_wh);
  k_prepack<<<1, 256, 0, stream>>>(ro_w1, 32, 2*2*32, pro_w1);
  k_prepack<<<1, 256, 0, stream>>>(ro_w2, 16, 1*1*32, pro_w2);

  // ---- encoders ----
  k_load<<<(cL+255)/256, 256, 0, stream>>>(ft, cap, ptl, load_b);
  k_path_l1<<<(cP*cD+255)/256, 256, 0, stream>>>(ft, fpk, fsz, pe_w1, pe_b1, s1);
  gemm_bias_act<4,2,1><<<cP/16, 32, 0, stream>>>(s1, cD, ppe_w2, pe_b2, hP, cD, cP);
  k_link_l1<<<(cL*cD+255)/256, 256, 0, stream>>>(cap, load_b, ldt, le_w1, le_b1, s1);
  gemm_bias_act<4,2,1><<<cL/16, 32, 0, stream>>>(s1, cD, ple_w2, le_b2, hL, cD, cL);
  k_dlm<<<(cN+255)/256, 256, 0, stream>>>(hL, ltn, dlm);
  k_dev_l1<<<(cN*cD+255)/256, 256, 0, stream>>>(nodes, dlm, de_w1, de_b1, s1);
  gemm_bias_act<4,2,1><<<cN/16, 32, 0, stream>>>(s1, cD, pde_w2, de_b2, hD, cD, cN);

  // ---- message-passing iterations ----
  for (int it = 0; it < cITER; ++it){
    k_copy_prev<<<(cP*cD+255)/256, 256, 0, stream>>>(hP, pss);
    for (int t = 0; t < cT; ++t)
      gru_fused<<<cP/16, 32, 0, stream>>>(nullptr, ltp, ntp, hL, hD, t,
                                          ppg_wx, ppg_wh, pgru_bx, pgru_bh,
                                          hP, pss, cP, 0);
    k_agg<<<(cL*cD+255)/256, 256, 0, stream>>>(pss, ptl, cK, agg, cL);
    gru_fused<<<cL/16, 32, 0, stream>>>(agg, nullptr, nullptr, nullptr, nullptr, 0,
                                        plg_wx, plg_wh, lgru_bx, lgru_bh,
                                        hL, nullptr, cL, 1);
    k_agg<<<(cN*cD+255)/256, 256, 0, stream>>>(pss, ptn, cK2, agg, cN);
    gru_fused<<<cN/16, 32, 0, stream>>>(agg, nullptr, nullptr, nullptr, nullptr, 0,
                                        pdg_wx, pdg_wh, dgru_bx, dgru_bh,
                                        hD, nullptr, cN, 1);
  }

  // ---- readout ----
  k_pssc<<<(cP*cT*cD+255)/256, 256, 0, stream>>>(pss, s1);
  gemm_bias_act<2,2,1><<<(cP*cT)/16, 32, 0, stream>>>(s1, cD, pro_w1, ro_b1, s2, 32, cP*cT);
  gemm_bias_act<1,1,1><<<(cP*cT)/16, 32, 0, stream>>>(s2, 32, pro_w2, ro_b2, s3, 16, cP*cT);
  k_final<<<(cP+255)/256, 256, 0, stream>>>(s3, ro_w3, ro_b3, cap, ltp, out);
}
